// GAT_36550171689609
// MI455X (gfx1250) — matrix operand order repaired and checked by reading, not compiled, not run on hardware
//
#include <hip/hip_runtime.h>
#include <hip/hip_bf16.h>

// ---------------------------------------------------------------------------
// Constants from the reference
// ---------------------------------------------------------------------------
constexpr int N_   = 25600;      // nodes
constexpr int F_   = 400;        // input features
constexpr int KP1  = 448;        // F_ padded to multiple of 64
constexpr int H_   = 64;         // hidden
constexpr int E_   = 1638400;    // edges
constexpr int G_   = 64;         // graphs

// gfx1250 async global->LDS path (guarded: falls back to sync copies if the
// toolchain does not declare these builtins)
#if defined(__gfx1250__) && __has_builtin(__builtin_amdgcn_global_load_async_to_lds_b128) && \
    __has_builtin(__builtin_amdgcn_s_wait_asynccnt)
#define USE_ASYNC_LDS 1
#else
#define USE_ASYNC_LDS 0
#endif

// ---------------------------------------------------------------------------
// Types for WMMA / async copies
// ---------------------------------------------------------------------------
typedef __attribute__((ext_vector_type(16))) __bf16 v16bf;
typedef __attribute__((ext_vector_type(8)))  float  v8f;
typedef int v4i_vec __attribute__((vector_size(4 * sizeof(int))));
typedef __attribute__((address_space(1))) v4i_vec* gptr_v4i;
typedef __attribute__((address_space(3))) v4i_vec* lptr_v4i;

union Frag {
    v16bf v;
    uint4 q[2];
};

__device__ __forceinline__ unsigned short f2bf(float f) {
    unsigned u = __float_as_uint(f);
    unsigned r = u + 0x7fffu + ((u >> 16) & 1u);   // round-to-nearest-even
    return (unsigned short)(r >> 16);
}

__device__ __forceinline__ unsigned fenc(float f) {   // order-preserving uint encode
    unsigned u = __float_as_uint(f);
    return (u & 0x80000000u) ? ~u : (u | 0x80000000u);
}
__device__ __forceinline__ float fdec(unsigned u) {
    u = (u & 0x80000000u) ? (u & 0x7fffffffu) : ~u;
    return __uint_as_float(u);
}
__device__ __forceinline__ float lrelu02(float x) { return x > 0.f ? x : 0.2f * x; }

// ---------------------------------------------------------------------------
// Conversion kernels (f32 -> bf16)
// ---------------------------------------------------------------------------
// A-side: pad columns: in [rows,K] -> out [rows,Kp], zero-fill k in [K,Kp)
__global__ void cvt2d_bf16(const float* __restrict__ in, unsigned short* __restrict__ out,
                           int rows, int K, int Kp) {
    size_t total = (size_t)rows * (size_t)Kp;
    size_t stride = (size_t)gridDim.x * blockDim.x;
    for (size_t idx = (size_t)blockIdx.x * blockDim.x + threadIdx.x; idx < total; idx += stride) {
        size_t r = idx / (size_t)Kp;
        int k = (int)(idx - r * (size_t)Kp);
        float f = (k < K) ? in[r * (size_t)K + k] : 0.f;
        out[idx] = f2bf(f);
    }
}

// A-side flat convert (no padding needed)
__global__ void cvt1d_bf16(const float* __restrict__ in, unsigned short* __restrict__ out,
                           size_t ntotal) {
    size_t stride = (size_t)gridDim.x * blockDim.x;
    for (size_t idx = (size_t)blockIdx.x * blockDim.x + threadIdx.x; idx < ntotal; idx += stride)
        out[idx] = f2bf(in[idx]);
}

// B-side: transpose + pad: in [K,64] row-major -> out [64,Kp] (column-major B),
// so GEMM LDS staging becomes contiguous 16B copies (async-to-LDS friendly).
__global__ void cvtB_bf16(const float* __restrict__ in, unsigned short* __restrict__ out,
                          int K, int Kp) {
    size_t total = (size_t)64 * (size_t)Kp;
    size_t stride = (size_t)gridDim.x * blockDim.x;
    for (size_t idx = (size_t)blockIdx.x * blockDim.x + threadIdx.x; idx < total; idx += stride) {
        int n = (int)(idx / (size_t)Kp);
        int k = (int)(idx - (size_t)n * (size_t)Kp);
        float f = (k < K) ? in[(size_t)k * 64 + n] : 0.f;
        out[idx] = f2bf(f);
    }
}

__global__ void zero_f32(float* __restrict__ p, size_t n) {
    size_t stride = (size_t)gridDim.x * blockDim.x;
    for (size_t i = (size_t)blockIdx.x * blockDim.x + threadIdx.x; i < n; i += stride)
        p[i] = 0.f;
}

// ---------------------------------------------------------------------------
// WMMA bf16 GEMM:  C[M,64] = A[M,Kp] @ B[Kp,64]
//   A: bf16 row-major [M,Kp]; Btg: bf16 COLUMN-major [64,Kp]; C: f32 [M,64]
// Block = 256 threads = 8 waves; each wave computes a 16x64 strip.
// B chunks (64x64) double-buffered in LDS via async global->LDS copies.
// ---------------------------------------------------------------------------
__global__ __launch_bounds__(256) void gemm_bf16(
    const unsigned short* __restrict__ A,
    const unsigned short* __restrict__ Btg,
    float* __restrict__ C,
    int M, int Kp)
{
    __shared__ unsigned short Bt[2][64][72];  // [buf][n][k], row stride 144B (16B aligned)

    const int lane  = threadIdx.x & 31;
    const int wave  = threadIdx.x >> 5;
    const int l16   = lane & 15;
    const int khalf = lane >> 4;
    const int rowBase = blockIdx.x * 128 + wave * 16;

    int rA = rowBase + l16;
    if (rA >= M) rA = M - 1;                 // clamp (keeps all waves in barriers)
    const unsigned short* Arow = A + (size_t)rA * (size_t)Kp;

    v8f acc0 = {}, acc1 = {}, acc2 = {}, acc3 = {};

    // stage one 64x64 B chunk into LDS buffer `buf` (two 16B ops per thread)
    auto stage = [&](int k0, int buf) {
#pragma unroll
        for (int rep = 0; rep < 2; ++rep) {
            int i = (int)threadIdx.x + rep * 256;     // 0..511
            int n = i >> 3;
            int kk8 = (i & 7) * 8;
            const unsigned short* src = Btg + (size_t)n * (size_t)Kp + k0 + kk8;
#if USE_ASYNC_LDS
            __builtin_amdgcn_global_load_async_to_lds_b128(
                (gptr_v4i)(src),
                (lptr_v4i)(&Bt[buf][n][kk8]),
                0, 0);
#else
            *(uint4*)(&Bt[buf][n][kk8]) = *(const uint4*)(src);
#endif
        }
    };

    const int nchunks = Kp >> 6;
    stage(0, 0);

    for (int c = 0; c < nchunks; ++c) {
        const int k0 = c << 6;
        const int buf = c & 1;

        if (c + 1 < nchunks) {
            stage(k0 + 64, buf ^ 1);
#if USE_ASYNC_LDS
            __builtin_amdgcn_s_wait_asynccnt(2);   // chunk c's copies complete
#endif
        } else {
#if USE_ASYNC_LDS
            __builtin_amdgcn_s_wait_asynccnt(0);
#endif
        }
        __syncthreads();

        if (k0 + 64 < Kp)                          // gfx1250 global_prefetch on next A chunk
            __builtin_prefetch(Arow + k0 + 64, 0, 0);

#pragma unroll
        for (int ks = 0; ks < 64; ks += 32) {
            // A fragment (16-bit A 16x32 layout): K = ks + khalf*8 + [0..8) and +16
            Frag a;
            const unsigned short* ap = Arow + k0 + ks + khalf * 8;
            a.q[0] = *(const uint4*)(ap);
            a.q[1] = *(const uint4*)(ap + 16);

#pragma unroll
            for (int t = 0; t < 4; ++t) {
                // B fragment (32x16): lane holds column n, K = khalf*8 + [0..8) and +16
                Frag b;
                const unsigned short* bp = &Bt[buf][t * 16 + l16][ks + khalf * 8];
                b.q[0] = *(const uint4*)(bp);
                b.q[1] = *(const uint4*)(bp + 16);

                v8f& acc = (t == 0) ? acc0 : (t == 1) ? acc1 : (t == 2) ? acc2 : acc3;
                acc = __builtin_amdgcn_wmma_f32_16x16x32_bf16(
                          false, a.v, false, b.v, (short)0, acc, false, false);
            }
        }
        __syncthreads();
    }

    if (rowBase < M) {
#pragma unroll
        for (int t = 0; t < 4; ++t) {
            v8f acc = (t == 0) ? acc0 : (t == 1) ? acc1 : (t == 2) ? acc2 : acc3;
            int col = t * 16 + l16;
#pragma unroll
            for (int j = 0; j < 8; ++j) {
                int row = rowBase + (khalf ? (8 + j) : j);
                if (row < M) C[(size_t)row * 64 + col] = acc[j];
            }
        }
    }
}

// ---------------------------------------------------------------------------
// Attention pipeline kernels
// ---------------------------------------------------------------------------
__global__ void scores_kernel(const float* __restrict__ h,
                              const float* __restrict__ a_src,
                              const float* __restrict__ a_dst,
                              float* __restrict__ ssrc, float* __restrict__ sdst, int n)
{
    int lane = threadIdx.x & 31;
    int node = (blockIdx.x * blockDim.x + threadIdx.x) >> 5;
    if (node >= n) return;
    const float* row = h + (size_t)node * 64;
    float v0 = row[lane], v1 = row[lane + 32];
    float s1 = v0 * a_src[lane] + v1 * a_src[lane + 32];
    float s2 = v0 * a_dst[lane] + v1 * a_dst[lane + 32];
#pragma unroll
    for (int off = 16; off > 0; off >>= 1) {
        s1 += __shfl_down(s1, off, 32);
        s2 += __shfl_down(s2, off, 32);
    }
    if (lane == 0) { ssrc[node] = s1; sdst[node] = s2; }
}

__global__ void node_init(const float* __restrict__ ssrc, const float* __restrict__ sdst,
                          unsigned* __restrict__ smax, float* __restrict__ denom, int n)
{
    int i = blockIdx.x * blockDim.x + threadIdx.x;
    if (i >= n) return;
    float self = lrelu02(ssrc[i] + sdst[i]);
    smax[i] = fenc(self);
    denom[i] = 0.f;
}

__global__ void edge_max(const int* __restrict__ src, const int* __restrict__ dst,
                         const float* __restrict__ ew,
                         const float* __restrict__ ssrc, const float* __restrict__ sdst,
                         unsigned* __restrict__ smax, int e)
{
    int i = blockIdx.x * blockDim.x + threadIdx.x;
    if (i >= e) return;
    if (ew[i] > 0.f) {
        int s = src[i], d = dst[i];
        float sc = lrelu02(ssrc[s] + sdst[d]);
        atomicMax(&smax[d], fenc(sc));
    }
}

__global__ void edge_exp(const int* __restrict__ src, const int* __restrict__ dst,
                         const float* __restrict__ ew,
                         const float* __restrict__ ssrc, const float* __restrict__ sdst,
                         const unsigned* __restrict__ smax,
                         float* __restrict__ exb, float* __restrict__ denom, int e)
{
    int i = blockIdx.x * blockDim.x + threadIdx.x;
    if (i >= e) return;
    float ex = 0.f;
    if (ew[i] > 0.f) {
        int s = src[i], d = dst[i];
        float sc = lrelu02(ssrc[s] + sdst[d]);
        ex = __expf(sc - fdec(smax[d]));
        atomicAdd(&denom[d], ex);
    }
    exb[i] = ex;
}

__global__ void node_self(const float* __restrict__ ssrc, const float* __restrict__ sdst,
                          const unsigned* __restrict__ smax,
                          float* __restrict__ denom, float* __restrict__ selfex, int n)
{
    int i = blockIdx.x * blockDim.x + threadIdx.x;
    if (i >= n) return;
    float self = lrelu02(ssrc[i] + sdst[i]);
    float es = __expf(self - fdec(smax[i]));
    selfex[i] = es;
    denom[i] += es;
}

// wave per edge: agg[d] += (ex/denom[d]) * h[s]   (2 f32 atomics per lane)
__global__ __launch_bounds__(256) void edge_agg(
    const int* __restrict__ src, const int* __restrict__ dst,
    const float* __restrict__ exb, const float* __restrict__ denom,
    const float* __restrict__ h, float* __restrict__ agg, int e)
{
    int lane = threadIdx.x & 31;
    int eid = (blockIdx.x * blockDim.x + threadIdx.x) >> 5;
    if (eid >= e) return;
    float ex = exb[eid];
    if (ex == 0.f) return;
    int s = src[eid], d = dst[eid];
    float al = ex / denom[d];
    const float* hs = h + (size_t)s * 64;
    float* ad = agg + (size_t)d * 64;
    atomicAdd(&ad[lane],      al * hs[lane]);
    atomicAdd(&ad[lane + 32], al * hs[lane + 32]);
}

__global__ void finalize_layer(const float* __restrict__ agg,
                               const float* __restrict__ selfex, const float* __restrict__ denom,
                               const float* __restrict__ h, const float* __restrict__ bias,
                               float slope, float* __restrict__ out, int n)
{
    size_t total = (size_t)n * 64;
    size_t stride = (size_t)gridDim.x * blockDim.x;
    for (size_t idx = (size_t)blockIdx.x * blockDim.x + threadIdx.x; idx < total; idx += stride) {
        size_t i = idx >> 6;
        int c = (int)(idx & 63);
        float v = agg[idx] + (selfex[i] / denom[i]) * h[idx] + bias[c];
        out[idx] = (v < 0.f) ? v * slope : v;
    }
}

__global__ void readout_final(const float* __restrict__ C3, const float* __restrict__ rb,
                              const float* __restrict__ lW, const float* __restrict__ lb,
                              float* __restrict__ out)
{
    int g = threadIdx.x;
    if (g < G_) {
        float acc = lb[0];
        for (int c = 0; c < 64; ++c)
            acc += (C3[g * 64 + c] + rb[c]) * lW[c];
        out[g] = acc;
    }
}

// ---------------------------------------------------------------------------
// Host-side orchestration
// ---------------------------------------------------------------------------
static constexpr size_t al256(size_t x) { return (x + 255) & ~(size_t)255; }

extern "C" void kernel_launch(void* const* d_in, const int* in_sizes, int n_in,
                              void* d_out, int out_size, void* d_ws, size_t ws_size,
                              hipStream_t stream) {
    const float* x   = (const float*)d_in[0];
    const float* ew  = (const float*)d_in[1];
    const float* W1  = (const float*)d_in[2];
    const float* a1s = (const float*)d_in[3];
    const float* a1d = (const float*)d_in[4];
    const float* b1  = (const float*)d_in[5];
    const float* W2  = (const float*)d_in[6];
    const float* a2s = (const float*)d_in[7];
    const float* a2d = (const float*)d_in[8];
    const float* b2  = (const float*)d_in[9];
    const float* rW  = (const float*)d_in[10];
    const float* rb  = (const float*)d_in[11];
    const float* lW  = (const float*)d_in[12];
    const float* lb  = (const float*)d_in[13];
    const int*   eix = (const int*)d_in[14];
    const int* esrc = eix;
    const int* edst = eix + E_;

    // workspace layout
    char* ws = (char*)d_ws;
    size_t off = 0;
    unsigned short* ABF = (unsigned short*)(ws + off); off = al256(off + (size_t)N_ * KP1 * 2);
    unsigned short* WB  = (unsigned short*)(ws + off); off = al256(off + (size_t)N_ * H_ * 2);
    float* Hb   = (float*)(ws + off); off = al256(off + (size_t)N_ * H_ * 4);
    float* OUTb = (float*)(ws + off); off = al256(off + (size_t)N_ * H_ * 4);
    float* AGG  = (float*)(ws + off); off = al256(off + (size_t)N_ * H_ * 4);
    float* EXB  = (float*)(ws + off); off = al256(off + (size_t)E_ * 4);
    float* SSRC = (float*)(ws + off); off = al256(off + (size_t)N_ * 4);
    float* SDST = (float*)(ws + off); off = al256(off + (size_t)N_ * 4);
    unsigned* SMAX = (unsigned*)(ws + off); off = al256(off + (size_t)N_ * 4);
    float* DEN  = (float*)(ws + off); off = al256(off + (size_t)N_ * 4);
    float* SELF = (float*)(ws + off); off = al256(off + (size_t)N_ * 4);
    (void)ws_size; (void)n_in; (void)in_sizes; (void)out_size;

    const int TB = 256;
    auto gsz = [](size_t total, int tb) { return (int)((total + tb - 1) / tb); };

    auto attention = [&](const float* h, const float* as, const float* ad, const float* bias,
                         float slope, float* out) {
        scores_kernel<<<N_ / 8, TB, 0, stream>>>(h, as, ad, SSRC, SDST, N_);
        node_init<<<gsz(N_, TB), TB, 0, stream>>>(SSRC, SDST, SMAX, DEN, N_);
        zero_f32<<<gsz((size_t)N_ * H_, TB), TB, 0, stream>>>(AGG, (size_t)N_ * H_);
        edge_max<<<gsz(E_, TB), TB, 0, stream>>>(esrc, edst, ew, SSRC, SDST, SMAX, E_);
        edge_exp<<<gsz(E_, TB), TB, 0, stream>>>(esrc, edst, ew, SSRC, SDST, SMAX, EXB, DEN, E_);
        node_self<<<gsz(N_, TB), TB, 0, stream>>>(SSRC, SDST, SMAX, DEN, SELF, N_);
        edge_agg<<<E_ / 8, TB, 0, stream>>>(esrc, edst, EXB, DEN, h, AGG, E_);
        finalize_layer<<<gsz((size_t)N_ * H_, TB), TB, 0, stream>>>(AGG, SELF, DEN, h, bias,
                                                                    slope, out, N_);
    };

    // ---- Layer 1: h = x @ W1, GAT attention, leaky_relu(0.01) ----
    cvt2d_bf16<<<gsz((size_t)N_ * KP1, TB), TB, 0, stream>>>(x, ABF, N_, F_, KP1);
    cvtB_bf16<<<gsz((size_t)64 * KP1, TB), TB, 0, stream>>>(W1, WB, F_, KP1);
    gemm_bf16<<<N_ / 128, TB, 0, stream>>>(ABF, WB, Hb, N_, KP1);
    attention(Hb, a1s, a1d, b1, 0.01f, OUTb);

    // ---- Layer 2: h = out1 @ W2, GAT attention (no activation) ----
    cvt1d_bf16<<<gsz((size_t)N_ * H_, TB), TB, 0, stream>>>(OUTb, ABF, (size_t)N_ * H_);
    cvtB_bf16<<<gsz((size_t)64 * H_, TB), TB, 0, stream>>>(W2, WB, H_, H_);
    gemm_bf16<<<N_ / 128, TB, 0, stream>>>(ABF, WB, Hb, N_, H_);
    attention(Hb, a2s, a2d, b2, 1.0f, OUTb);

    // ---- Readout: g = reshape(out2)[64,25600] @ rW[25600,64] (+rb), y = g@lW + lb ----
    cvt1d_bf16<<<gsz((size_t)N_ * H_, TB), TB, 0, stream>>>(OUTb, ABF, (size_t)N_ * H_);
    cvtB_bf16<<<gsz((size_t)64 * (N_ * H_ / G_), TB), TB, 0, stream>>>(rW, WB, N_ * H_ / G_,
                                                                      N_ * H_ / G_);
    gemm_bf16<<<1, TB, 0, stream>>>(ABF, WB, Hb, G_, N_ * H_ / G_);
    readout_final<<<1, 64, 0, stream>>>(Hb, rb, lW, lb, (float*)d_out);
}
